// Person_To_Group_82351702934098
// MI455X (gfx1250) — matrix operand order repaired and checked by reading, not compiled, not run on hardware
//
#include <hip/hip_runtime.h>
#include <hip/hip_bf16.h>

typedef __bf16 bf16;
typedef __attribute__((ext_vector_type(16))) __bf16 v16bf;
typedef __attribute__((ext_vector_type(8)))  float  v8f;

#define D 512          // feature dim (reference uses 512)
#define GCLUST 2       // clusters
#define ROWS_PER_WAVE 16
#define BLOCK 256      // 8 waves (wave32)
#define ROWS_PER_WG (8 * ROWS_PER_WAVE)   // 128 rows per workgroup
#define KCOL 16        // columns per lane in the sum pass (D/32)

// ---------------------------------------------------------------------------
// Wave-level assignment of 16 points via v_wmma_f32_16x16x32_bf16.
//   A (16x32 bf16) = normalized centers, row r = center (r&1)  -> rows 2..15
//       of D are garbage and ignored; loaded unconditionally (no divergence).
//   B (32x16 bf16) = 16 rows of x as columns (column n = point n).
//   D row 0 = dot(c0, point n) in VGPR0 lane n; row 1 = dot(c1, .) in VGPR1.
// Returns 16-bit mask: bit n = 1 <=> point (rowbase+n) belongs to cluster 1.
// ---------------------------------------------------------------------------
__device__ __forceinline__ unsigned wave_assign_mask(const float* __restrict__ x,
                                                     size_t rowbase,
                                                     const bf16* s_cbf) {
  const int lane = threadIdx.x & 31;
  const int lrow = lane & 15;     // A row / B column (point index in tile)
  const int hi   = lane >> 4;     // lane half selects K sub-range
  const bf16*  cb = s_cbf + (lrow & 1) * D;          // center for A row lrow
  const float* xr = x + (rowbase + (size_t)lrow) * D; // point for B col lrow

  v8f acc = {0.f, 0.f, 0.f, 0.f, 0.f, 0.f, 0.f, 0.f};

  #pragma unroll 4
  for (int kb = 0; kb < D; kb += 32) {
    __builtin_prefetch(xr + kb + 128);     // global_prefetch_b8, near scope
    // A fragment (16-bit A 16x32): lane<16 K=kb..kb+15;
    //                              lane>=16 K=kb+16..kb+31
    const int ka = kb + hi * 16;
    v16bf a;
    #pragma unroll
    for (int j = 0; j < 8; ++j) {
      a[j]     = cb[ka + j];
      a[8 + j] = cb[ka + 8 + j];
    }
    // B fragment (16-bit B 32x16): lane<16 K=kb..kb+15; lane>=16 K=kb+16..kb+31
    const int kx = kb + hi * 16;
    v16bf b;
    #pragma unroll
    for (int j = 0; j < 4; ++j) {
      float4 p = *(const float4*)(xr + kx + 4 * j);
      b[4 * j]     = (bf16)p.x;  b[4 * j + 1] = (bf16)p.y;
      b[4 * j + 2] = (bf16)p.z;  b[4 * j + 3] = (bf16)p.w;
    }
    acc = __builtin_amdgcn_wmma_f32_16x16x32_bf16(false, a, false, b,
                                                  (short)0, acc, false, false);
  }
  // Lane n (n<16): acc[0] = dot(c0, point n), acc[1] = dot(c1, point n).
  const unsigned long long bl = __ballot(acc[1] > acc[0]);
  return (unsigned)bl & 0xFFFFu;
}

// ---------------------------------------------------------------------------
// Per-wave fused cluster sum: register partials, then 2 LDS atomics/column.
// s_sums layout: [cluster][D].
// ---------------------------------------------------------------------------
__device__ __forceinline__ void wave_accumulate(const float* __restrict__ x,
                                                size_t rowbase, unsigned m,
                                                float* s_sums) {
  const int lane = threadIdx.x & 31;
  float ta[KCOL], t1[KCOL];
  #pragma unroll
  for (int kk = 0; kk < KCOL; ++kk) { ta[kk] = 0.f; t1[kk] = 0.f; }
  for (int r = 0; r < ROWS_PER_WAVE; ++r) {
    const float idf = (float)((m >> r) & 1u);
    const float* xr = x + (rowbase + r) * D;
    #pragma unroll
    for (int kk = 0; kk < KCOL; ++kk) {
      const float v = xr[kk * 32 + lane];
      ta[kk] += v;
      t1[kk] += v * idf;      // branchless: only cluster-1 rows contribute
    }
  }
  #pragma unroll
  for (int kk = 0; kk < KCOL; ++kk) {
    const int k = kk * 32 + lane;
    atomicAdd(&s_sums[D + k], t1[kk]);            // ds_add_f32
    atomicAdd(&s_sums[k],     ta[kk] - t1[kk]);
  }
}

// ---------------------------------------------------------------------------
// init: centers = {x[0], x[M/2]}, zero sums/counts, flag=0, normalized bf16.
// ---------------------------------------------------------------------------
__global__ void p2g_init(const float* __restrict__ x, float* centers,
                         float* sums, float* counts, int* flag, bf16* cbf,
                         int M) {
  __shared__ float red[BLOCK];
  const int tid = threadIdx.x;
  const size_t i1 = (size_t)(M / 2);
  for (int k = tid; k < D; k += BLOCK) {
    centers[k]     = x[k];
    centers[D + k] = x[i1 * D + k];
  }
  for (int i = tid; i < GCLUST * D; i += BLOCK) sums[i] = 0.f;
  if (tid < GCLUST) counts[tid] = 0.f;
  if (tid == 0) *flag = 0;
  __syncthreads();
  for (int g = 0; g < GCLUST; ++g) {
    float a = 0.f;
    for (int k = tid; k < D; k += BLOCK) { float v = centers[g * D + k]; a += v * v; }
    red[tid] = a; __syncthreads();
    for (int s = BLOCK / 2; s > 0; s >>= 1) {
      if (tid < s) red[tid] += red[tid + s];
      __syncthreads();
    }
    const float inv = 1.f / fmaxf(sqrtf(red[0]), 1e-12f);
    __syncthreads();
    for (int k = tid; k < D; k += BLOCK)
      cbf[g * D + k] = (bf16)(centers[g * D + k] * inv);
    __syncthreads();
  }
}

// ---------------------------------------------------------------------------
// assign + fused cluster-sum: one streaming pass over x.
// grid = M / ROWS_PER_WG, block = 256.
// ---------------------------------------------------------------------------
__global__ void p2g_assign(const float* __restrict__ x,
                           const bf16* __restrict__ cbf,
                           float* sums, float* counts,
                           const int* __restrict__ flag) {
  if (*flag) return;
  __shared__ bf16  s_cbf[GCLUST * D];
  __shared__ float s_sums[GCLUST * D];
  __shared__ float s_cnt1;
  const int tid = threadIdx.x;
  for (int i = tid; i < GCLUST * D; i += BLOCK) { s_cbf[i] = cbf[i]; s_sums[i] = 0.f; }
  if (tid == 0) s_cnt1 = 0.f;
  __syncthreads();

  const int wave = tid >> 5, lane = tid & 31;
  const size_t rowbase = (size_t)blockIdx.x * ROWS_PER_WG + wave * ROWS_PER_WAVE;
  const unsigned m = wave_assign_mask(x, rowbase, s_cbf);
  wave_accumulate(x, rowbase, m, s_sums);
  if (lane == 0) atomicAdd(&s_cnt1, (float)__popc(m));
  __syncthreads();

  for (int i = tid; i < GCLUST * D; i += BLOCK) atomicAdd(&sums[i], s_sums[i]);
  if (tid == 0) {
    atomicAdd(&counts[1], s_cnt1);
    atomicAdd(&counts[0], (float)ROWS_PER_WG - s_cnt1);
  }
}

// ---------------------------------------------------------------------------
// update: new = sums/counts (keep old if empty), shift^2 convergence test,
// write centers + normalized bf16 centers, zero sums/counts. One workgroup.
// ---------------------------------------------------------------------------
__global__ void p2g_update(float* centers, float* sums, float* counts,
                           int* flag, bf16* cbf) {
  if (*flag) return;
  __shared__ float s_new[GCLUST * D];
  __shared__ float red[BLOCK];
  __shared__ float s_shift;
  const int tid = threadIdx.x;
  if (tid == 0) s_shift = 0.f;
  const float c[GCLUST] = {counts[0], counts[1]};
  __syncthreads();

  for (int g = 0; g < GCLUST; ++g) {
    float a = 0.f;
    for (int k = tid; k < D; k += BLOCK) {
      const float old = centers[g * D + k];
      const float nv  = (c[g] > 0.f) ? sums[g * D + k] / fmaxf(c[g], 1.f) : old;
      s_new[g * D + k] = nv;
      const float dv = nv - old;
      a += dv * dv;
    }
    red[tid] = a; __syncthreads();
    for (int s = BLOCK / 2; s > 0; s >>= 1) {
      if (tid < s) red[tid] += red[tid + s];
      __syncthreads();
    }
    if (tid == 0) s_shift += sqrtf(red[0]);
    __syncthreads();
  }
  if (tid == 0) {
    const float sh2 = s_shift * s_shift;
    if (sh2 < 1e-4f) *flag = 1;      // converged: later iterations no-op
  }
  for (int g = 0; g < GCLUST; ++g) {
    float a = 0.f;
    for (int k = tid; k < D; k += BLOCK) { const float v = s_new[g * D + k]; a += v * v; }
    red[tid] = a; __syncthreads();
    for (int s = BLOCK / 2; s > 0; s >>= 1) {
      if (tid < s) red[tid] += red[tid + s];
      __syncthreads();
    }
    const float inv = 1.f / fmaxf(sqrtf(red[0]), 1e-12f);
    __syncthreads();
    for (int k = tid; k < D; k += BLOCK) {
      const float v = s_new[g * D + k];
      centers[g * D + k] = v;
      cbf[g * D + k]     = (bf16)(v * inv);
      sums[g * D + k]    = 0.f;
    }
    __syncthreads();
  }
  if (tid < GCLUST) counts[tid] = 0.f;
}

__global__ void p2g_zero(float* out, int n) {
  const int i = blockIdx.x * BLOCK + threadIdx.x;
  if (i < n) out[i] = 0.f;
}

// ---------------------------------------------------------------------------
// final scatter: re-assign with final centers (WMMA), per-batch LDS partial
// sums, global atomic add into out[b, g, :].  grid = M / ROWS_PER_WG.
// ---------------------------------------------------------------------------
__global__ void p2g_scatter(const float* __restrict__ x,
                            const bf16* __restrict__ cbf,
                            float* __restrict__ out, int N) {
  __shared__ bf16  s_cbf[GCLUST * D];
  __shared__ float s_sums[GCLUST * D];
  const int tid = threadIdx.x;
  for (int i = tid; i < GCLUST * D; i += BLOCK) { s_cbf[i] = cbf[i]; s_sums[i] = 0.f; }
  __syncthreads();

  const int wave = tid >> 5;
  const size_t wgbase  = (size_t)blockIdx.x * ROWS_PER_WG;     // N % 128 == 0
  const size_t rowbase = wgbase + wave * ROWS_PER_WAVE;
  const unsigned m = wave_assign_mask(x, rowbase, s_cbf);
  wave_accumulate(x, rowbase, m, s_sums);
  __syncthreads();

  const size_t batch = wgbase / (size_t)N;
  float* ob = out + batch * (GCLUST * D);
  for (int i = tid; i < GCLUST * D; i += BLOCK) atomicAdd(&ob[i], s_sums[i]);
}

// ---------------------------------------------------------------------------
extern "C" void kernel_launch(void* const* d_in, const int* in_sizes, int n_in,
                              void* d_out, int out_size, void* d_ws, size_t ws_size,
                              hipStream_t stream) {
  const float* x = (const float*)d_in[0];
  float* out = (float*)d_out;

  const int M = in_sizes[0] / D;            // 131072
  const int B = out_size / (GCLUST * D);    // 32
  const int N = M / B;                      // 4096

  // workspace layout (floats)
  float* ws      = (float*)d_ws;
  float* centers = ws;                       // 2*512
  float* sums    = ws + GCLUST * D;          // 2*512
  float* counts  = ws + 2 * GCLUST * D;      // 2
  int*   flag    = (int*)(ws + 2 * GCLUST * D + 2);
  bf16*  cbf     = (bf16*)(ws + 2 * GCLUST * D + 4);  // 2*512 bf16

  p2g_init<<<1, BLOCK, 0, stream>>>(x, centers, sums, counts, flag, cbf, M);

  const int nwg = M / ROWS_PER_WG;          // 1024
  for (int it = 0; it < 100; ++it) {        // ITER_LIMIT; device-side early-out
    p2g_assign<<<nwg, BLOCK, 0, stream>>>(x, cbf, sums, counts, flag);
    p2g_update<<<1, BLOCK, 0, stream>>>(centers, sums, counts, flag, cbf);
  }

  p2g_zero<<<(out_size + BLOCK - 1) / BLOCK, BLOCK, 0, stream>>>(out, out_size);
  p2g_scatter<<<nwg, BLOCK, 0, stream>>>(x, cbf, out, N);
}
